// MoeMLP_17008070492396
// MI455X (gfx1250) — matrix operand order repaired and checked by reading, not compiled, not run on hardware
//
#include <hip/hip_runtime.h>
#include <hip/hip_bf16.h>
#include <math.h>

#define NTOK   8192
#define NE     8
#define NEMBD  1024
#define DFFN   2048
#define TW     16384
#define TILE_M 128
#define MAX_TILES 136   // ceil(2*NTOK/128) + NE

typedef __attribute__((ext_vector_type(16))) __bf16 v16bf;
typedef __attribute__((ext_vector_type(8)))  __bf16 v8bf;
typedef __attribute__((ext_vector_type(8)))  float  v8f;
typedef __attribute__((ext_vector_type(4)))  int    v4i;

// ---- CDNA5 async global->LDS path (guarded; falls back to VGPR copies) ----
#if defined(__has_builtin)
#if __has_builtin(__builtin_amdgcn_global_load_async_to_lds_b128)
#define HAS_ASYNC_LDS 1
#endif
#endif

#ifdef HAS_ASYNC_LDS
#define GLD_ASYNC_B128(gsrc, ldst)                                             \
    __builtin_amdgcn_global_load_async_to_lds_b128(                            \
        (__attribute__((address_space(1))) v4i*)(gsrc),                        \
        (__attribute__((address_space(3))) v4i*)(ldst), 0, 0)
#if __has_builtin(__builtin_amdgcn_s_wait_asynccnt)
#define WAIT_ASYNC(n) __builtin_amdgcn_s_wait_asynccnt(n)
#else
#define WAIT_ASYNC(n) asm volatile("s_wait_asynccnt %0" ::"i"(n) : "memory")
#endif
#define NABUF 3
#else
#define NABUF 2
#endif

// ---- workspace layout (bytes) ----
#define WS_COUNTS 0                              // 8 ints
#define WS_CW     256                            // NTOK*8 floats
#define WS_LIST   (WS_CW + NTOK*NE*4)            // NE*NTOK ints
#define WS_NT     (WS_LIST + NE*NTOK*4)          // 1 int (numTiles)
#define WS_TEXP   (WS_NT + 256)                  // MAX_TILES ints
#define WS_TROW   (WS_TEXP + 1024)               // MAX_TILES ints
#define WS_ZP     (WS_TROW + 1024)               // 1024 floats
#define WS_PP     (WS_ZP + 4096)                 // 1024*8 floats
#define WS_H      (WS_PP + NE*1024*4)            // bf16 h: MAX_TILES*128 x DFFN

__device__ __forceinline__ unsigned pk2(float a, float b) {
    unsigned short ua = __builtin_bit_cast(unsigned short, (__bf16)a);
    unsigned short ub = __builtin_bit_cast(unsigned short, (__bf16)b);
    return (unsigned)ua | ((unsigned)ub << 16);
}
__device__ __forceinline__ v16bf cat16(v8bf lo, v8bf hi) {
    return __builtin_shufflevector(lo, hi, 0,1,2,3,4,5,6,7,8,9,10,11,12,13,14,15);
}

// ---------------- zero output + counters ----------------
__global__ void k_zero(float* __restrict__ out, int* __restrict__ counts) {
    size_t i = (size_t)blockIdx.x * blockDim.x + threadIdx.x;
    if (i < (size_t)NTOK * NEMBD) out[i] = 0.0f;
    if (blockIdx.x == 0 && threadIdx.x < NE) counts[threadIdx.x] = 0;
}

// ---------------- router: logits, softmax, top-2, lists ----------------
__global__ __launch_bounds__(256) void k_router(
    const float* __restrict__ x, const float* __restrict__ wr,
    float* __restrict__ cw, int* __restrict__ counts, int* __restrict__ list,
    float* __restrict__ zpart, float* __restrict__ ppart)
{
    __shared__ float wrs[NE * NEMBD];          // 32 KB
    __shared__ float zb[8];
    __shared__ float pb[8][NE];
    for (int i = threadIdx.x; i < NE * NEMBD; i += 256) wrs[i] = wr[i];
    __syncthreads();

    int wave = threadIdx.x >> 5, lane = threadIdx.x & 31;
    int t = blockIdx.x * 8 + wave;

    float acc[NE];
    #pragma unroll
    for (int e = 0; e < NE; ++e) acc[e] = 0.0f;
    const float* xp = x + (size_t)t * NEMBD;
    #pragma unroll 4
    for (int i = 0; i < NEMBD / 32; ++i) {
        float xv = xp[lane + i * 32];
        #pragma unroll
        for (int e = 0; e < NE; ++e) acc[e] += xv * wrs[e * NEMBD + lane + i * 32];
    }
    #pragma unroll
    for (int off = 16; off > 0; off >>= 1)
        #pragma unroll
        for (int e = 0; e < NE; ++e) acc[e] += __shfl_xor(acc[e], off, 32);

    float m = acc[0];
    #pragma unroll
    for (int e = 1; e < NE; ++e) m = fmaxf(m, acc[e]);
    float p[NE], s = 0.0f;
    #pragma unroll
    for (int e = 0; e < NE; ++e) { p[e] = expf(acc[e] - m); s += p[e]; }
    float inv = 1.0f / s;
    #pragma unroll
    for (int e = 0; e < NE; ++e) p[e] *= inv;
    float lse = m + logf(s);
    float z = lse * lse;

    int e1 = 0;
    #pragma unroll
    for (int e = 1; e < NE; ++e) if (p[e] > p[e1]) e1 = e;
    int e2 = (e1 == 0) ? 1 : 0;
    #pragma unroll
    for (int e = 0; e < NE; ++e) if (e != e1 && p[e] > p[e2]) e2 = e;
    float sw = p[e1] + p[e2];
    float w1n = p[e1] / sw, w2n = p[e2] / sw;

    if (lane == 0) {
        #pragma unroll
        for (int e = 0; e < NE; ++e)
            cw[(size_t)t * NE + e] = (e == e1) ? w1n : ((e == e2) ? w2n : 0.0f);
        int pos = atomicAdd(&counts[e1], 1); list[e1 * NTOK + pos] = t;
        pos     = atomicAdd(&counts[e2], 1); list[e2 * NTOK + pos] = t;
        zb[wave] = z;
        #pragma unroll
        for (int e = 0; e < NE; ++e) pb[wave][e] = p[e];
    }
    __syncthreads();
    if (threadIdx.x == 0) {
        float zs = 0.0f;
        for (int w = 0; w < 8; ++w) zs += zb[w];
        zpart[blockIdx.x] = zs;
    }
    if (threadIdx.x < NE) {
        float ps = 0.0f;
        for (int w = 0; w < 8; ++w) ps += pb[w][threadIdx.x];
        ppart[blockIdx.x * NE + threadIdx.x] = ps;
    }
}

// ---------------- build tile map ----------------
__global__ void k_tilemap(const int* __restrict__ counts, int* __restrict__ nt,
                          int* __restrict__ texp, int* __restrict__ trow) {
    if (threadIdx.x == 0 && blockIdx.x == 0) {
        int n = 0;
        for (int e = 0; e < NE; ++e) {
            int tl = (counts[e] + TILE_M - 1) / TILE_M;
            for (int i = 0; i < tl; ++i) { texp[n] = e; trow[n] = i * TILE_M; n++; }
        }
        *nt = n;
    }
}

// ---------------- aux losses (deterministic fixed-order reduce) ----------------
__global__ void k_finalize(const float* __restrict__ zpart, const float* __restrict__ ppart,
                           const int* __restrict__ counts, float* __restrict__ tail) {
    __shared__ float ps[NE];
    if (threadIdx.x == 0) {
        float zs = 0.0f;
        for (int i = 0; i < 1024; ++i) zs += zpart[i];
        tail[0] = zs / (float)NTOK;
    }
    if (threadIdx.x < NE) {
        float s = 0.0f;
        for (int i = 0; i < 1024; ++i) s += ppart[i * NE + threadIdx.x];
        ps[threadIdx.x] = s / (float)NTOK;
    }
    __syncthreads();
    if (threadIdx.x == 0) {
        float lb = 0.0f;
        for (int e = 0; e < NE; ++e) {
            float fi = (float)counts[e] / (float)(2 * NTOK);
            tail[2 + e] = fi;
            lb += fi * ps[e];
        }
        tail[1] = (float)NE * lb;
    }
}

// ---------------- grouped GEMM 1: h = gelu(Xg @ W1_e) ----------------
// software-pipelined: reg-staged global loads 2 steps ahead, double-buffered
// LDS, one barrier per K-step.
__global__ __launch_bounds__(256) void k_gemm1(
    const float* __restrict__ x, const float* __restrict__ w1,
    const int* __restrict__ nt, const int* __restrict__ texp, const int* __restrict__ trow,
    const int* __restrict__ counts, const int* __restrict__ list,
    __bf16* __restrict__ h)
{
    int tile = blockIdx.x;
    if (tile >= *nt) return;
    int e = texp[tile], rs = trow[tile], cnt = counts[e];

    __shared__ int tok[TILE_M];
    __shared__ alignas(32) __bf16 Ab[2][TILE_M][48];
    __shared__ alignas(32) __bf16 Bt[2][128][48];

    int tid = threadIdx.x;
    if (tid < TILE_M) {
        int r = rs + tid;
        tok[tid] = (r < cnt) ? list[e * NTOK + r] : -1;
    }
    __syncthreads();

    int lane = tid & 31, wave = tid >> 5;
    int wm = wave & 1, wn = wave >> 1;          // 2 M-groups x 4 N-groups
    int l15 = lane & 15, khalf = lane >> 4;

    v8f c[4][2];
    #pragma unroll
    for (int mi = 0; mi < 4; ++mi)
        #pragma unroll
        for (int ni = 0; ni < 2; ++ni) c[mi][ni] = {};

    const int n0 = e * DFFN + blockIdx.y * 128;
    int lm = tid >> 1;
    int lj = (tid & 1) * 16;
    int bk = tid & 31;
    int bc = (tid >> 5) * 16;
    int tkn = tok[lm];

    float4 ra[4], rb[4];
    auto loadA = [&](int kk) {
        if (tkn >= 0) {
            const float4* s = (const float4*)(x + (size_t)tkn * NEMBD + kk + lj);
            #pragma unroll
            for (int q = 0; q < 4; ++q) ra[q] = s[q];
        } else {
            float4 z = {0.f, 0.f, 0.f, 0.f};
            #pragma unroll
            for (int q = 0; q < 4; ++q) ra[q] = z;
        }
    };
    auto loadB = [&](int kk) {
        const float4* s = (const float4*)(w1 + (size_t)(kk + bk) * TW + n0 + bc);
        #pragma unroll
        for (int q = 0; q < 4; ++q) rb[q] = s[q];
    };
    auto storeA = [&](int p) {
        #pragma unroll
        for (int q = 0; q < 4; ++q)
            *(uint2*)&Ab[p][lm][lj + q * 4] = (uint2){ pk2(ra[q].x, ra[q].y), pk2(ra[q].z, ra[q].w) };
    };
    auto storeB = [&](int p) {
        #pragma unroll
        for (int q = 0; q < 4; ++q) {
            Bt[p][bc + q * 4 + 0][bk] = (__bf16)rb[q].x;
            Bt[p][bc + q * 4 + 1][bk] = (__bf16)rb[q].y;
            Bt[p][bc + q * 4 + 2][bk] = (__bf16)rb[q].z;
            Bt[p][bc + q * 4 + 3][bk] = (__bf16)rb[q].w;
        }
    };
    auto computeT = [&](int p) {
        v16bf bf2[2];
        #pragma unroll
        for (int ni = 0; ni < 2; ++ni) {
            int n = wn * 32 + ni * 16 + l15;
            v8bf blo = *(const v8bf*)&Bt[p][n][khalf * 8];
            v8bf bhi = *(const v8bf*)&Bt[p][n][16 + khalf * 8];
            bf2[ni] = cat16(blo, bhi);
        }
        #pragma unroll
        for (int mi = 0; mi < 4; ++mi) {
            int mrow = wm * 64 + mi * 16 + l15;
            v8bf alo = *(const v8bf*)&Ab[p][mrow][khalf * 8];
            v8bf ahi = *(const v8bf*)&Ab[p][mrow][16 + khalf * 8];
            v16bf af = cat16(alo, ahi);
            c[mi][0] = __builtin_amdgcn_wmma_f32_16x16x32_bf16(false, af, false, bf2[0],
                                                               (short)0, c[mi][0], false, false);
            c[mi][1] = __builtin_amdgcn_wmma_f32_16x16x32_bf16(false, af, false, bf2[1],
                                                               (short)0, c[mi][1], false, false);
        }
    };

    const int NSTEP = NEMBD / 32;  // 32
    loadA(0); loadB(0);
    storeA(0); storeB(0);
    loadA(32); loadB(32);
    __syncthreads();
    for (int i = 0; i < NSTEP; ++i) {
        int p = i & 1;
        computeT(p);
        if (i + 1 < NSTEP) { storeA(p ^ 1); storeB(p ^ 1); }
        if (i + 2 < NSTEP) { loadA((i + 2) * 32); loadB((i + 2) * 32); }
        __syncthreads();
    }

    // epilogue: exact gelu, store bf16 h
    size_t hbase = (size_t)tile * TILE_M;
    int nb = blockIdx.y * 128 + wn * 32;
    #pragma unroll
    for (int mi = 0; mi < 4; ++mi)
        #pragma unroll
        for (int ni = 0; ni < 2; ++ni) {
            int ncol = nb + ni * 16 + l15;
            #pragma unroll
            for (int r = 0; r < 8; ++r) {
                int mrow = wm * 64 + mi * 16 + khalf * 8 + r;
                float v = c[mi][ni][r];
                float g = 0.5f * v * (1.0f + erff(v * 0.70710678118f));
                h[(hbase + mrow) * DFFN + ncol] = (__bf16)g;
            }
        }
}

// ---------------- grouped GEMM 2: out += cw * (h @ W2_e) ----------------
// A tile is bf16 already: copied with GLOBAL_LOAD_ASYNC_TO_LDS_B128 (triple-
// buffered, ASYNCcnt pipelined) when the toolchain exposes the builtin.
__global__ __launch_bounds__(256) void k_gemm2(
    const __bf16* __restrict__ h, const float* __restrict__ w2,
    const float* __restrict__ cw,
    const int* __restrict__ nt, const int* __restrict__ texp, const int* __restrict__ trow,
    const int* __restrict__ counts, const int* __restrict__ list,
    float* __restrict__ out)
{
    int tile = blockIdx.x;
    if (tile >= *nt) return;
    int e = texp[tile], rs = trow[tile], cnt = counts[e];

    __shared__ int   tokL[TILE_M];
    __shared__ float cwL[TILE_M];
    __shared__ alignas(32) __bf16 Ab[NABUF][TILE_M][48];
    __shared__ alignas(32) __bf16 Bt[2][128][48];

    int tid = threadIdx.x;
    if (tid < TILE_M) {
        int r = rs + tid;
        int tk = (r < cnt) ? list[e * NTOK + r] : -1;
        tokL[tid] = tk;
        cwL[tid] = (tk >= 0) ? cw[(size_t)tk * NE + e] : 0.0f;
    }
    __syncthreads();

    int lane = tid & 31, wave = tid >> 5;
    int wm = wave & 1, wn = wave >> 1;
    int l15 = lane & 15, khalf = lane >> 4;

    v8f c[4][2];
    #pragma unroll
    for (int mi = 0; mi < 4; ++mi)
        #pragma unroll
        for (int ni = 0; ni < 2; ++ni) c[mi][ni] = {};

    const int n0 = blockIdx.y * 128;
    int lm = tid >> 1;
    int lj = (tid & 1) * 16;
    int bk = tid & 31;
    int bc = (tid >> 5) * 16;
    const __bf16* hrow = h + ((size_t)tile * TILE_M + lm) * DFFN + lj;

    float4 rb[4];
    auto loadB = [&](int step) {
        const float4* s = (const float4*)(w2 + (size_t)(e * DFFN + step * 32 + bk) * NEMBD + n0 + bc);
        #pragma unroll
        for (int q = 0; q < 4; ++q) rb[q] = s[q];
    };
    auto storeB = [&](int p) {
        #pragma unroll
        for (int q = 0; q < 4; ++q) {
            Bt[p][bc + q * 4 + 0][bk] = (__bf16)rb[q].x;
            Bt[p][bc + q * 4 + 1][bk] = (__bf16)rb[q].y;
            Bt[p][bc + q * 4 + 2][bk] = (__bf16)rb[q].z;
            Bt[p][bc + q * 4 + 3][bk] = (__bf16)rb[q].w;
        }
    };
    auto computeT = [&](int pa, int pb) {
        v16bf bf2[2];
        #pragma unroll
        for (int ni = 0; ni < 2; ++ni) {
            int n = wn * 32 + ni * 16 + l15;
            v8bf blo = *(const v8bf*)&Bt[pb][n][khalf * 8];
            v8bf bhi = *(const v8bf*)&Bt[pb][n][16 + khalf * 8];
            bf2[ni] = cat16(blo, bhi);
        }
        #pragma unroll
        for (int mi = 0; mi < 4; ++mi) {
            int mrow = wm * 64 + mi * 16 + l15;
            v8bf alo = *(const v8bf*)&Ab[pa][mrow][khalf * 8];
            v8bf ahi = *(const v8bf*)&Ab[pa][mrow][16 + khalf * 8];
            v16bf af = cat16(alo, ahi);
            c[mi][0] = __builtin_amdgcn_wmma_f32_16x16x32_bf16(false, af, false, bf2[0],
                                                               (short)0, c[mi][0], false, false);
            c[mi][1] = __builtin_amdgcn_wmma_f32_16x16x32_bf16(false, af, false, bf2[1],
                                                               (short)0, c[mi][1], false, false);
        }
    };

    const int NSTEP = DFFN / 32;  // 64

#ifdef HAS_ASYNC_LDS
    auto asyncA = [&](int step, int buf) {
        const __bf16* g = hrow + step * 32;
        GLD_ASYNC_B128(g,     &Ab[buf][lm][lj]);
        GLD_ASYNC_B128(g + 8, &Ab[buf][lm][lj + 8]);
    };
    // prologue: A tiles 0,1 in flight; B tile 0 staged, B tile 1 in regs
    asyncA(0, 0);
    asyncA(1, 1);
    loadB(0); storeB(0);
    loadB(1);
    WAIT_ASYNC(2);            // tile 0 landed (in-order per thread)
    __syncthreads();
    for (int i = 0; i < NSTEP; ++i) {
        int pa = i % 3, pb = i & 1;
        if (i + 2 < NSTEP) asyncA(i + 2, (i + 2) % 3);  // buffer last read at i-1
        computeT(pa, pb);
        if (i + 1 < NSTEP) storeB(pb ^ 1);
        if (i + 2 < NSTEP) { loadB(i + 2); WAIT_ASYNC(2); }  // retire A(i+1)
        else               { WAIT_ASYNC(0); }
        __syncthreads();
    }
#else
    uint4 ua[2];
    auto loadA = [&](int step) {
        const uint4* s = (const uint4*)(hrow + step * 32);
        ua[0] = s[0]; ua[1] = s[1];
    };
    auto storeA = [&](int p) {
        *(uint4*)&Ab[p][lm][lj]     = ua[0];
        *(uint4*)&Ab[p][lm][lj + 8] = ua[1];
    };
    loadA(0); loadB(0);
    storeA(0); storeB(0);
    loadA(1); loadB(1);
    __syncthreads();
    for (int i = 0; i < NSTEP; ++i) {
        int p = i & 1;
        computeT(p, p);
        if (i + 1 < NSTEP) { storeA(p ^ 1); storeB(p ^ 1); }
        if (i + 2 < NSTEP) { loadA(i + 2); loadB(i + 2); }
        __syncthreads();
    }
#endif

    // epilogue: scale by combine weight, atomic scatter-add to output
    int nb = n0 + wn * 32;
    #pragma unroll
    for (int mi = 0; mi < 4; ++mi)
        #pragma unroll
        for (int ni = 0; ni < 2; ++ni) {
            int ncol = nb + ni * 16 + l15;
            #pragma unroll
            for (int r = 0; r < 8; ++r) {
                int mrow = wm * 64 + mi * 16 + khalf * 8 + r;
                int tk = tokL[mrow];
                if (tk >= 0)
                    atomicAdd(&out[(size_t)tk * NEMBD + ncol], c[mi][ni][r] * cwL[mrow]);
            }
        }
}

extern "C" void kernel_launch(void* const* d_in, const int* in_sizes, int n_in,
                              void* d_out, int out_size, void* d_ws, size_t ws_size,
                              hipStream_t stream) {
    const float* x  = (const float*)d_in[0];
    const float* wr = (const float*)d_in[1];
    const float* w1 = (const float*)d_in[2];
    const float* w2 = (const float*)d_in[3];
    float* out = (float*)d_out;

    char* ws = (char*)d_ws;
    int*    counts = (int*)(ws + WS_COUNTS);
    float*  cw     = (float*)(ws + WS_CW);
    int*    list   = (int*)(ws + WS_LIST);
    int*    nt     = (int*)(ws + WS_NT);
    int*    texp   = (int*)(ws + WS_TEXP);
    int*    trow   = (int*)(ws + WS_TROW);
    float*  zp     = (float*)(ws + WS_ZP);
    float*  pp     = (float*)(ws + WS_PP);
    __bf16* h      = (__bf16*)(ws + WS_H);

    k_zero<<<(NTOK * NEMBD) / 256, 256, 0, stream>>>(out, counts);
    k_router<<<NTOK / 8, 256, 0, stream>>>(x, wr, cw, counts, list, zp, pp);
    k_tilemap<<<1, 32, 0, stream>>>(counts, nt, texp, trow);
    k_finalize<<<1, 32, 0, stream>>>(zp, pp, counts, out + (size_t)NTOK * NEMBD);
    k_gemm1<<<dim3(MAX_TILES, DFFN / 128), 256, 0, stream>>>(x, w1, nt, texp, trow, counts, list, h);
    k_gemm2<<<dim3(MAX_TILES, NEMBD / 128), 256, 0, stream>>>(h, w2, cw, nt, texp, trow, counts, list, out);
}
